// MSAttention_80418967650699
// MI455X (gfx1250) — compile-verified, matrix operand order checked by reading, not run on hardware
//
#include <hip/hip_runtime.h>
#include <hip/hip_bf16.h>
#include <math.h>

// Problem constants (reference: B=8, H=W=56, C=384, 8 heads, hd=48, SR=8)
#define NB   8
#define NN   3136
#define CCH  384

typedef __bf16 bf16_t;
typedef __bf16 bf16x8  __attribute__((ext_vector_type(8)));
typedef __bf16 bf16x16 __attribute__((ext_vector_type(16)));
typedef float  f32x8   __attribute__((ext_vector_type(8)));

// ---------------------------------------------------------------------------
// WMMA helpers.  A-fragment (16x32 bf16, ISA 7.12.2): lane L -> row M=L%16,
// halves 0..7 -> K = kb..kb+7, halves 8..15 -> K = kb+16..kb+23 with
// kb = (L&16)?8:0.  B-operand is staged K-major per output column so the same
// loader serves both.  C/D (16x16 f32): VGPR r, lane L -> M = r + 8*(L/16),
// N = L%16.
// ---------------------------------------------------------------------------
__device__ __forceinline__ bf16x16 lds_frag(const bf16_t* base, int ld, int lane) {
  const int r  = lane & 15;
  const int kb = (lane & 16) ? 8 : 0;
  const bf16_t* p = base + r * ld + kb;
  bf16x8 lo = *(const bf16x8*)(p);
  bf16x8 hi = *(const bf16x8*)(p + 16);
  bf16x16 f;
#pragma unroll
  for (int i = 0; i < 8; ++i) { f[i] = lo[i]; f[i + 8] = hi[i]; }
  return f;
}

__device__ __forceinline__ f32x8 wmma_bf16(bf16x16 a, bf16x16 b, f32x8 c) {
  return __builtin_amdgcn_wmma_f32_16x16x32_bf16(false, a, false, b,
                                                 (short)0, c, false, false);
}

// CDNA5 async global->LDS copy (GLOBAL_LOAD_ASYNC_TO_LDS_B128, ASYNCcnt).
// LDS destination is the low 32 bits of the generic pointer (LDS offset).
__device__ __forceinline__ void async_b128(void* lds_dst, const void* gsrc) {
  unsigned l = (unsigned)(size_t)lds_dst;
  asm volatile("global_load_async_to_lds_b128 %0, %1, off"
               :: "v"(l), "v"(gsrc)
               : "memory");
}
__device__ __forceinline__ void wait_async() {
  asm volatile("s_wait_asynccnt 0" ::: "memory");
}

// ---------------------------------------------------------------------------
// Elementwise converters
// ---------------------------------------------------------------------------
__global__ void k_cvt_bf16(const float* __restrict__ src, bf16_t* __restrict__ dst, long n) {
  long i = (long)blockIdx.x * blockDim.x + threadIdx.x;
  if (i < n) dst[i] = (bf16_t)src[i];
}

// dst[n][k] = bf16(src[k][n])   (src is [K][Nc] row-major f32)
__global__ void k_transpose_cvt(const float* __restrict__ src, bf16_t* __restrict__ dst,
                                int K, int Nc) {
  long i = (long)blockIdx.x * blockDim.x + threadIdx.x;
  long tot = (long)K * Nc;
  if (i >= tot) return;
  int n = (int)(i / K);
  int k = (int)(i % K);
  dst[i] = (bf16_t)src[(long)k * Nc + n];
}

__global__ void k_zero_bf16(bf16_t* __restrict__ p, long n) {
  long i = (long)blockIdx.x * blockDim.x + threadIdx.x;
  if (i < n) p[i] = (bf16_t)0.0f;
}

// ---------------------------------------------------------------------------
// Generic 128x64 bf16 WMMA GEMM, K-step 64 (8 WMMA per barrier pair).
// A:[M][K] bf16, Bt:[Nout][K] bf16.  256 threads = 8 waves, wave(wm,wn) owns
// a 32x32 block (4 accumulator tiles).  Full tiles are staged with async
// global->LDS copies; ragged M tiles fall back to guarded VGPR staging.
// mode 0: outQ bf16 = acc*scale (ld 384)           (Q projection, scale folded)
// mode 1: outF f32  = acc + bias[col] (ld ldout)   (final projection)
// mode 2: KV scatter: col<192 -> Kp[b,h,m,d] bf16 (padded [Nip][64]),
//                     col>=192 -> Vf[b,h,m,d] f32
// ---------------------------------------------------------------------------
__global__ __launch_bounds__(256) void k_gemm(
    const bf16_t* __restrict__ A, const bf16_t* __restrict__ Bt,
    int M, int K, int mode, float scale,
    bf16_t* __restrict__ outQ,
    const float* __restrict__ bias, float* __restrict__ outF, int ldout,
    bf16_t* __restrict__ Kp, float* __restrict__ Vf, int Ni, int Nip) {
  __shared__ bf16_t As[128 * 64];
  __shared__ bf16_t Bs[64 * 64];
  const int t = threadIdx.x;
  const int lane = t & 31, wid = t >> 5;
  const int wm = wid & 3, wn = wid >> 2;
  const int m0 = blockIdx.x * 128;
  const int n0 = blockIdx.y * 64;
  const bool fullM = (m0 + 128) <= M;

  f32x8 zero8 = {};
  f32x8 acc[2][2] = {{zero8, zero8}, {zero8, zero8}};

  for (int k0 = 0; k0 < K; k0 += 64) {
    __syncthreads();
    if (fullM) {                                // async stage A (128x64)
#pragma unroll
      for (int it = 0; it < 4; ++it) {
        int q = t + it * 256;
        int row = q >> 3, cs = (q & 7) * 8;
        async_b128(As + row * 64 + cs, A + (long)(m0 + row) * K + k0 + cs);
      }
    } else {                                    // ragged tail: guarded staging
#pragma unroll
      for (int it = 0; it < 4; ++it) {
        int q = t + it * 256;
        int row = q >> 3, cs = (q & 7) * 8;
        int rg = m0 + row;
        bf16x8 v = {};
        if (rg < M) v = *(const bf16x8*)(A + (long)rg * K + k0 + cs);
        *(bf16x8*)(As + row * 64 + cs) = v;
      }
    }
#pragma unroll
    for (int it = 0; it < 2; ++it) {            // async stage Bt (64x64)
      int q = t + it * 256;
      int row = q >> 3, cs = (q & 7) * 8;
      async_b128(Bs + row * 64 + cs, Bt + (long)(n0 + row) * K + k0 + cs);
    }
    if (k0 + 64 < K) {                          // CDNA5 global_prefetch_b8
      int row = t >> 3, cs = (t & 7) * 8;
      int rg = m0 + row;
      if (rg < M) __builtin_prefetch(A + (long)rg * K + k0 + 64 + cs, 0, 1);
    }
    wait_async();
    __syncthreads();

#pragma unroll
    for (int kk = 0; kk < 64; kk += 32) {
      bf16x16 a0 = lds_frag(As + (wm * 32) * 64 + kk, 64, lane);
      bf16x16 a1 = lds_frag(As + (wm * 32 + 16) * 64 + kk, 64, lane);
      bf16x16 b0 = lds_frag(Bs + (wn * 32) * 64 + kk, 64, lane);
      bf16x16 b1 = lds_frag(Bs + (wn * 32 + 16) * 64 + kk, 64, lane);
      acc[0][0] = wmma_bf16(a0, b0, acc[0][0]);
      acc[0][1] = wmma_bf16(a0, b1, acc[0][1]);
      acc[1][0] = wmma_bf16(a1, b0, acc[1][0]);
      acc[1][1] = wmma_bf16(a1, b1, acc[1][1]);
    }
  }

  const int rofs = (lane & 16) ? 8 : 0;
  const int cofs = lane & 15;
#pragma unroll
  for (int ti = 0; ti < 2; ++ti)
#pragma unroll
    for (int tj = 0; tj < 2; ++tj)
#pragma unroll
      for (int r = 0; r < 8; ++r) {
        int rg = m0 + wm * 32 + ti * 16 + r + rofs;
        if (rg >= M) continue;
        int col = n0 + wn * 32 + tj * 16 + cofs;
        float v = acc[ti][tj][r];
        if (mode == 0) {
          outQ[(long)rg * CCH + col] = (bf16_t)(v * scale);
        } else if (mode == 1) {
          outF[(long)rg * ldout + col] = v + bias[col];
        } else {
          int b = rg / Ni, m = rg % Ni;
          if (col < 192) {
            int h = col / 48, d = col % 48;
            Kp[(((long)b * 4 + h) * Nip + m) * 64 + d] = (bf16_t)v;
          } else {
            int c2 = col - 192;
            int h = c2 / 48, d = c2 % 48;
            Vf[(((long)b * 4 + h) * Ni + m) * 48 + d] = v;
          }
        }
      }
}

// ---------------------------------------------------------------------------
// Stride==kernel conv as gathered-A GEMM (K-step 64).  A row = (b,py,px)
// patch; K index kk = (dy*ks+dx)*384 + cin.  384 % 64 == 0, so every 64-chunk
// stays inside one (dy,dx) pixel -> contiguous 16B loads from xb.  Full-row
// tiles use async global->LDS staging.  Epilogue: y f32 + bias.
// ---------------------------------------------------------------------------
__global__ __launch_bounds__(256) void k_gemm_conv(
    const bf16_t* __restrict__ xb, const bf16_t* __restrict__ Bt,
    const float* __restrict__ bias, float* __restrict__ y,
    int Ni, int Wi, int ks, int K) {
  __shared__ bf16_t As[128 * 64];
  __shared__ bf16_t Bs[64 * 64];
  const int t = threadIdx.x;
  const int lane = t & 31, wid = t >> 5;
  const int wm = wid & 3, wn = wid >> 2;
  const int m0 = blockIdx.x * 128;
  const int n0 = blockIdx.y * 64;
  const int M = NB * Ni;
  const bool fullM = (m0 + 128) <= M;

  f32x8 zero8 = {};
  f32x8 acc[2][2] = {{zero8, zero8}, {zero8, zero8}};

  for (int k0 = 0; k0 < K; k0 += 64) {
    __syncthreads();
    int pix = k0 / CCH;
    int c0  = k0 % CCH;
    int dy = pix / ks, dx = pix % ks;
#pragma unroll
    for (int it = 0; it < 4; ++it) {
      int q = t + it * 256;
      int row = q >> 3, cs = (q & 7) * 8;
      int rg = m0 + row;
      if (rg < M) {
        int b = rg / Ni, tok = rg % Ni;
        int py = tok / Wi, px = tok % Wi;
        int iy = py * ks + dy, ix = px * ks + dx;
        const bf16_t* src = xb + ((long)b * NN + iy * 56 + ix) * CCH + c0 + cs;
        if (fullM) {
          async_b128(As + row * 64 + cs, src);
        } else {
          *(bf16x8*)(As + row * 64 + cs) = *(const bf16x8*)(src);
        }
      } else {
        bf16x8 v = {};
        *(bf16x8*)(As + row * 64 + cs) = v;
      }
    }
#pragma unroll
    for (int it = 0; it < 2; ++it) {
      int q = t + it * 256;
      int row = q >> 3, cs = (q & 7) * 8;
      async_b128(Bs + row * 64 + cs, Bt + (long)(n0 + row) * K + k0 + cs);
    }
    wait_async();
    __syncthreads();

#pragma unroll
    for (int kk = 0; kk < 64; kk += 32) {
      bf16x16 a0 = lds_frag(As + (wm * 32) * 64 + kk, 64, lane);
      bf16x16 a1 = lds_frag(As + (wm * 32 + 16) * 64 + kk, 64, lane);
      bf16x16 b0 = lds_frag(Bs + (wn * 32) * 64 + kk, 64, lane);
      bf16x16 b1 = lds_frag(Bs + (wn * 32 + 16) * 64 + kk, 64, lane);
      acc[0][0] = wmma_bf16(a0, b0, acc[0][0]);
      acc[0][1] = wmma_bf16(a0, b1, acc[0][1]);
      acc[1][0] = wmma_bf16(a1, b0, acc[1][0]);
      acc[1][1] = wmma_bf16(a1, b1, acc[1][1]);
    }
  }

  const int rofs = (lane & 16) ? 8 : 0;
  const int cofs = lane & 15;
#pragma unroll
  for (int ti = 0; ti < 2; ++ti)
#pragma unroll
    for (int tj = 0; tj < 2; ++tj)
#pragma unroll
      for (int r = 0; r < 8; ++r) {
        int rg = m0 + wm * 32 + ti * 16 + r + rofs;
        if (rg >= M) continue;
        int col = n0 + wn * 32 + tj * 16 + cofs;
        y[(long)rg * CCH + col] = acc[ti][tj][r] + bias[col];
      }
}

// ---------------------------------------------------------------------------
// Rowwise LayerNorm (population var) + exact GELU, f32 -> bf16.  One block per
// 384-wide row; 128 threads x 3 elements.
// ---------------------------------------------------------------------------
__global__ __launch_bounds__(128) void k_ln_gelu(
    const float* __restrict__ y, const float* __restrict__ g,
    const float* __restrict__ bb, bf16_t* __restrict__ out) {
  __shared__ float red[8];
  const int row = blockIdx.x;
  const int t = threadIdx.x;
  const float* xr = y + (long)row * CCH;
  float x0 = xr[t], x1 = xr[t + 128], x2 = xr[t + 256];
  float s  = x0 + x1 + x2;
  float sq = x0 * x0 + x1 * x1 + x2 * x2;
#pragma unroll
  for (int off = 16; off > 0; off >>= 1) {
    s  += __shfl_xor(s, off);
    sq += __shfl_xor(sq, off);
  }
  int wid = t >> 5;
  if ((t & 31) == 0) { red[wid] = s; red[wid + 4] = sq; }
  __syncthreads();
  float S  = red[0] + red[1] + red[2] + red[3];
  float SQ = red[4] + red[5] + red[6] + red[7];
  float mean = S * (1.0f / 384.0f);
  float var  = SQ * (1.0f / 384.0f) - mean * mean;
  float inv  = rsqrtf(var + 1e-5f);
  bf16_t* orow = out + (long)row * CCH;
#pragma unroll
  for (int i = 0; i < 3; ++i) {
    int c = t + i * 128;
    float v = (xr[c] - mean) * inv * g[c] + bb[c];
    orow[c] = (bf16_t)(0.5f * v * (1.0f + erff(v * 0.70710678118654752f)));
  }
}

// ---------------------------------------------------------------------------
// Depthwise 3x3 SAME conv enhancement of V (channel c2 = h*48+d over the
// Hi x Wi token grid), added to V, written TRANSPOSED as Vt[b,h,d,m] bf16 so
// the PV WMMA B-operand is contiguous in m.
// ---------------------------------------------------------------------------
__global__ void k_local_conv(
    const float* __restrict__ Vf, const float* __restrict__ w,
    const float* __restrict__ bias, bf16_t* __restrict__ Vt,
    int Ni, int Hi, int Wi, int Nip) {
  long idx = (long)blockIdx.x * blockDim.x + threadIdx.x;
  long tot = (long)NB * 4 * Ni * 48;
  if (idx >= tot) return;
  int d = (int)(idx % 48);
  int m = (int)((idx / 48) % Ni);
  int h = (int)((idx / ((long)48 * Ni)) % 4);
  int b = (int)(idx / ((long)48 * Ni * 4));
  int yy = m / Wi, xx = m % Wi;
  int c2 = h * 48 + d;
  const float* vb = Vf + (((long)b * 4 + h) * Ni) * 48 + d;
  float a = bias[c2];
#pragma unroll
  for (int dy = 0; dy < 3; ++dy) {
    int iy = yy + dy - 1;
    if (iy < 0 || iy >= Hi) continue;
#pragma unroll
    for (int dx = 0; dx < 3; ++dx) {
      int ix = xx + dx - 1;
      if (ix < 0 || ix >= Wi) continue;
      a += vb[(long)(iy * Wi + ix) * 48] * w[(dy * 3 + dx) * 192 + c2];
    }
  }
  float vout = vb[(long)m * 48] + a;
  Vt[(((long)b * 4 + h) * 48 + d) * Nip + m] = (bf16_t)vout;
}

// ---------------------------------------------------------------------------
// Fused attention (flash style): block = (b, h, 64 q-rows), 4 waves x 16 rows.
// Q tile staged once (d padded 48->64 with zeros).  Per 32-key chunk (staged
// with async global->LDS copies):
//   S = Q Kc^T      -> 4 wmma (2 m-tiles x 2 K-steps)
//   online softmax  -> shfl_xor row reductions, running (m,l) per row
//   P -> LDS (A-frag relayout) -> O += P Vc  (3 wmma)
// Padded key columns masked to -1e30; final O/l written straight into the
// concat (proj input) buffer as bf16.  Scale is pre-folded into Q.
// ---------------------------------------------------------------------------
__global__ __launch_bounds__(128) void k_attn(
    const bf16_t* __restrict__ Qs, const bf16_t* __restrict__ Kp,
    const bf16_t* __restrict__ Vt, bf16_t* __restrict__ cat,
    int Ni, int Nip, int qheadbase, int catoff) {
  __shared__ bf16_t Qsh[64 * 64];
  __shared__ bf16_t Ksh[32 * 64];
  __shared__ bf16_t Vsh[48 * 32];
  __shared__ bf16_t Psh[4 * 16 * 32];

  const int t = threadIdx.x;
  const int lane = t & 31, w = t >> 5;
  const int n0 = blockIdx.x * 64;
  const int h  = blockIdx.y;
  const int b  = blockIdx.z;
  const int qoff = (qheadbase + h) * 48;

  {  // stage Q tile [64 rows][64 d] with zero pad d>=48
    int row = t >> 1, seg = (t & 1) * 32;
    const bf16_t* src = Qs + ((long)b * NN + n0 + row) * CCH + qoff;
    bf16_t* dst = Qsh + row * 64 + seg;
#pragma unroll
    for (int c = 0; c < 32; ++c) {
      int d = seg + c;
      dst[c] = (d < 48) ? src[d] : (bf16_t)0.0f;
    }
  }

  f32x8 zero8 = {};
  f32x8 o0 = zero8, o1 = zero8, o2 = zero8;
  float mrun[8], lrun[8];
#pragma unroll
  for (int r = 0; r < 8; ++r) { mrun[r] = -1e30f; lrun[r] = 0.0f; }

  const bf16_t* Kbase = Kp + (((long)b * 4 + h) * Nip) * 64;
  const bf16_t* Vbase = Vt + (((long)b * 4 + h) * 48) * Nip;
  bf16_t* Pw = Psh + w * 16 * 32;
  const int rofs = (lane & 16) ? 8 : 0;
  const int cofs = lane & 15;

  const int nchunks = Nip / 32;
  for (int cidx = 0; cidx < nchunks; ++cidx) {
    const int m0 = cidx * 32;
    __syncthreads();
    {  // async stage K chunk [32 m][64 d]
      int row = t >> 2, seg = (t & 3) * 16;
      const bf16_t* src = Kbase + (long)(m0 + row) * 64 + seg;
      async_b128(Ksh + row * 64 + seg,     src);
      async_b128(Ksh + row * 64 + seg + 8, src + 8);
    }
    if (t < 96) {  // async stage V^T chunk [48 d][32 m]
      int row = t >> 1, seg = (t & 1) * 16;
      const bf16_t* src = Vbase + (long)row * Nip + m0 + seg;
      async_b128(Vsh + row * 32 + seg,     src);
      async_b128(Vsh + row * 32 + seg + 8, src + 8);
    }
    wait_async();
    __syncthreads();

    bf16x16 aq0 = lds_frag(Qsh + (w * 16) * 64, 64, lane);
    bf16x16 aq1 = lds_frag(Qsh + (w * 16) * 64 + 32, 64, lane);
    f32x8 s0 = zero8, s1 = zero8;
    s0 = wmma_bf16(aq0, lds_frag(Ksh, 64, lane), s0);
    s0 = wmma_bf16(aq1, lds_frag(Ksh + 32, 64, lane), s0);
    s1 = wmma_bf16(aq0, lds_frag(Ksh + 16 * 64, 64, lane), s1);
    s1 = wmma_bf16(aq1, lds_frag(Ksh + 16 * 64 + 32, 64, lane), s1);

    const int col0 = m0 + cofs, col1 = m0 + 16 + cofs;
#pragma unroll
    for (int r = 0; r < 8; ++r) {
      float a0 = (col0 < Ni) ? s0[r] : -1e30f;
      float a1 = (col1 < Ni) ? s1[r] : -1e30f;
      float mx = fmaxf(a0, a1);
#pragma unroll
      for (int off = 1; off < 16; off <<= 1) mx = fmaxf(mx, __shfl_xor(mx, off));
      float mnew  = fmaxf(mrun[r], mx);
      float alpha = __expf(mrun[r] - mnew);
      mrun[r] = mnew;
      float p0 = __expf(a0 - mnew);
      float p1 = __expf(a1 - mnew);
      float ps = p0 + p1;
#pragma unroll
      for (int off = 1; off < 16; off <<= 1) ps += __shfl_xor(ps, off);
      lrun[r] = lrun[r] * alpha + ps;
      o0[r] *= alpha; o1[r] *= alpha; o2[r] *= alpha;
      Pw[(r + rofs) * 32 + cofs]      = (bf16_t)p0;   // D-layout -> A-layout
      Pw[(r + rofs) * 32 + 16 + cofs] = (bf16_t)p1;
    }
    asm volatile("s_wait_dscnt 0" ::: "memory");       // wave-private LDS RAW
    bf16x16 ap = lds_frag(Pw, 32, lane);
    o0 = wmma_bf16(ap, lds_frag(Vsh, 32, lane), o0);
    o1 = wmma_bf16(ap, lds_frag(Vsh + 16 * 32, 32, lane), o1);
    o2 = wmma_bf16(ap, lds_frag(Vsh + 32 * 32, 32, lane), o2);
  }

#pragma unroll
  for (int r = 0; r < 8; ++r) {
    int n = n0 + w * 16 + r + rofs;
    float invl = 1.0f / lrun[r];
    long base = ((long)b * NN + n) * 576 + catoff + h * 48;
    cat[base + cofs]      = (bf16_t)(o0[r] * invl);
    cat[base + 16 + cofs] = (bf16_t)(o1[r] * invl);
    cat[base + 32 + cofs] = (bf16_t)(o2[r] * invl);
  }
}

// ---------------------------------------------------------------------------
extern "C" void kernel_launch(void* const* d_in, const int* in_sizes, int n_in,
                              void* d_out, int out_size, void* d_ws, size_t ws_size,
                              hipStream_t stream) {
  (void)in_sizes; (void)n_in; (void)out_size; (void)ws_size;
  const float* x      = (const float*)d_in[0];
  const float* Wq     = (const float*)d_in[3];
  const float* Wkv[3] = {(const float*)d_in[4], (const float*)d_in[5], (const float*)d_in[6]};
  const float* srw[3] = {(const float*)d_in[7], (const float*)d_in[9], (const float*)d_in[11]};
  const float* srb[3] = {(const float*)d_in[8], (const float*)d_in[10], (const float*)d_in[12]};
  const float* g[3]   = {(const float*)d_in[13], (const float*)d_in[15], (const float*)d_in[17]};
  const float* be[3]  = {(const float*)d_in[14], (const float*)d_in[16], (const float*)d_in[18]};
  const float* lcw[3] = {(const float*)d_in[19], (const float*)d_in[21], (const float*)d_in[23]};
  const float* lcb[3] = {(const float*)d_in[20], (const float*)d_in[22], (const float*)d_in[24]};
  const float* projw  = (const float*)d_in[25];
  const float* projb  = (const float*)d_in[26];
  float* out = (float*)d_out;

  char* ws = (char*)d_ws;
  size_t off = 0;
  auto alloc = [&](size_t bytes) -> void* {
    off = (off + 255) & ~(size_t)255;
    void* p = ws + off;
    off += bytes;
    return p;
  };

  const long MQ = (long)NB * NN;                    // 25088
  bf16_t* xb   = (bf16_t*)alloc(MQ * CCH * 2);
  bf16_t* Qs   = (bf16_t*)alloc(MQ * CCH * 2);
  bf16_t* cat  = (bf16_t*)alloc(MQ * 576 * 2);
  bf16_t* Wqt  = (bf16_t*)alloc((long)CCH * CCH * 2);
  bf16_t* Wkvt[3];
  for (int i = 0; i < 3; ++i) Wkvt[i] = (bf16_t*)alloc((long)CCH * CCH * 2);
  const int KcA[3] = {24576, 6144, 1536};
  bf16_t* srwt[3];
  for (int i = 0; i < 3; ++i) srwt[i] = (bf16_t*)alloc((long)KcA[i] * CCH * 2);
  bf16_t* pwt  = (bf16_t*)alloc(576L * CCH * 2);
  float*  ybuf = (float*)alloc(8L * 784 * CCH * 4);
  bf16_t* xbi  = (bf16_t*)alloc(8L * 784 * CCH * 2);
  bf16_t* Kp   = (bf16_t*)alloc(8L * 4 * 800 * 64 * 2);
  float*  Vf   = (float*)alloc(8L * 4 * 784 * 48 * 4);
  bf16_t* Vt   = (bf16_t*)alloc(8L * 4 * 48 * 800 * 2);

  // --- precision conversion -------------------------------------------------
  {
    long n = MQ * CCH;
    k_cvt_bf16<<<(int)((n + 255) / 256), 256, 0, stream>>>(x, xb, n);
  }
  k_transpose_cvt<<<(CCH * CCH + 255) / 256, 256, 0, stream>>>(Wq, Wqt, CCH, CCH);
  for (int i = 0; i < 3; ++i) {
    k_transpose_cvt<<<(CCH * CCH + 255) / 256, 256, 0, stream>>>(Wkv[i], Wkvt[i], CCH, CCH);
    long n = (long)KcA[i] * CCH;
    k_transpose_cvt<<<(int)((n + 255) / 256), 256, 0, stream>>>(srw[i], srwt[i], KcA[i], CCH);
  }
  k_transpose_cvt<<<(576 * CCH + 255) / 256, 256, 0, stream>>>(projw, pwt, 576, CCH);

  // --- Q projection (scale folded in) --------------------------------------
  const float scale = 1.0f / sqrtf(48.0f);
  k_gemm<<<dim3(196, 6), 256, 0, stream>>>(xb, Wqt, (int)MQ, CCH, 0, scale,
                                           Qs, nullptr, nullptr, 0,
                                           nullptr, nullptr, 0, 0);

  // --- three SR branches ----------------------------------------------------
  const int NiA[3]  = {49, 196, 784};
  const int NipA[3] = {64, 224, 800};
  const int HiA[3]  = {7, 14, 28};
  const int ksA[3]  = {8, 4, 2};
  const int qbA[3]  = {0, 4, 4};
  const int coA[3]  = {0, 192, 384};

  for (int i = 0; i < 3; ++i) {
    const int Ni = NiA[i], Nip = NipA[i], Hi = HiA[i];
    const int M = NB * Ni;
    // zero padded K / V^T so pad lanes contribute exact zeros
    {
      long nk = 8L * 4 * Nip * 64;
      k_zero_bf16<<<(int)((nk + 255) / 256), 256, 0, stream>>>(Kp, nk);
      long nv = 8L * 4 * 48 * Nip;
      k_zero_bf16<<<(int)((nv + 255) / 256), 256, 0, stream>>>(Vt, nv);
    }
    // patchify conv as GEMM -> y (f32)
    k_gemm_conv<<<dim3((M + 127) / 128, 6), 256, 0, stream>>>(
        xb, srwt[i], srb[i], ybuf, Ni, Hi, ksA[i], KcA[i]);
    // LN + GELU -> bf16 branch tokens
    k_ln_gelu<<<M, 128, 0, stream>>>(ybuf, g[i], be[i], xbi);
    // KV projection with scatter into padded K and f32 V
    k_gemm<<<dim3((M + 127) / 128, 6), 256, 0, stream>>>(
        xbi, Wkvt[i], M, CCH, 2, 1.0f,
        nullptr, nullptr, nullptr, 0, Kp, Vf, Ni, Nip);
    // depthwise 3x3 enhancement of V, write V^T bf16
    {
      long n = (long)NB * 4 * Ni * 48;
      k_local_conv<<<(int)((n + 255) / 256), 256, 0, stream>>>(
          Vf, lcw[i], lcb[i], Vt, Ni, Hi, Hi, Nip);
    }
    // fused flash attention -> concat buffer
    k_attn<<<dim3(NN / 64, 4, NB), 128, 0, stream>>>(
        Qs, Kp, Vt, cat, Ni, Nip, qbA[i], coA[i]);
  }

  // --- output projection ----------------------------------------------------
  k_gemm<<<dim3(196, 6), 256, 0, stream>>>(cat, pwt, (int)MQ, 576, 1, 1.0f,
                                           nullptr, projb, out, CCH,
                                           nullptr, nullptr, 0, 0);
}
